// Lstmcell_89799176225033
// MI455X (gfx1250) — matrix operand order repaired and checked by reading, not compiled, not run on hardware
//
#include <hip/hip_runtime.h>

// ---------------------------------------------------------------------------
// 2-layer LSTM (1 -> 51 -> 1), N=1024 seqs, T=2048 steps, for MI455X gfx1250.
//
// Strategy: one wave32 per 16-sequence batch tile (64 waves total).
//   gates(16 x 208) = A(208 x 64) * B(64 x 16) per step via
//   v_wmma_f32_16x16x32_f16 (13 gate tiles x 2 K-frags = 26 WMMA / step).
//   A rows (gate' = 4*h + g interleave; gate' 204..207 = layer-2 gates):
//     K 0..50  = W_hh1 (or W_ih2 for layer-2 rows)
//     K 51     = W_ih1 column      (x_t folded into the matmul)
//     K 52     = W_hh2             (h2 recurrence folded in)
//     K 53     = combined bias     (B row 53 == 1.0)
//   Layer 2 runs retimed one step behind (it needs the *current* h1, which is
//   exactly what the next iteration's B fragment carries) -> zero extra cost.
//   Gate interleave makes i,f,g,o of one h land in 4 consecutive VGPRs of the
//   same lane in the D fragment -> activations are fully lane-local.
//   B rebuild per step = 13 __shfl_xor(,16) + per-slot selects.
// ---------------------------------------------------------------------------

typedef __attribute__((ext_vector_type(16))) _Float16 v16h;
typedef __attribute__((ext_vector_type(8)))  float    v8f;
typedef __attribute__((ext_vector_type(8)))  unsigned int v8u;

constexpr int Hn    = 51;    // hidden size
constexpr int Nn    = 1024;  // batch
constexpr int Tn    = 2048;  // time steps
constexpr int NTILE = 13;    // 208 gate' columns / 16
constexpr int NFRAG = 26;    // 13 tiles x 2 K-fragments

__device__ __forceinline__ float tanh_fast(float x) {
#if __has_builtin(__builtin_amdgcn_tanhf)
  return __builtin_amdgcn_tanhf(x);          // gfx1250 v_tanh_f32
#else
  float ax = __builtin_fabsf(x);
  float e  = __builtin_amdgcn_exp2f(ax * -2.885390081777927f);  // exp(-2|x|)
  float t  = (1.0f - e) * __builtin_amdgcn_rcpf(1.0f + e);
  return __builtin_copysignf(t, x);
#endif
}
__device__ __forceinline__ float sig_fast(float x) {
  return __builtin_fmaf(0.5f, tanh_fast(0.5f * x), 0.5f);
}
__device__ __forceinline__ unsigned pkh2(float lo, float hi) {
  auto p = __builtin_amdgcn_cvt_pkrtz(lo, hi);   // __fp16 ext_vector(2)
  return __builtin_bit_cast(unsigned, p);
}

// ---------------------------------------------------------------------------
// Prep: pack A matrix (208 x 64, f16) into WMMA A-fragment register layout.
// 16-bit A 16x32 layout: M = lane%16 for all lanes;
//   lanes 0-15 : slots 0-7 -> K=0..15 (2/VGPR; contiguous K order, same as B)
//   lanes 16-31: slots 0-7 -> K=16..31
// apack layout: [frag(=2*j+k)][lane][slot] as u32 (f16x2), frag K base = 32k.
// ---------------------------------------------------------------------------
__global__ void lstm_pack(const float* __restrict__ Wih1, const float* __restrict__ Whh1,
                          const float* __restrict__ bih1, const float* __restrict__ bhh1,
                          const float* __restrict__ Wih2, const float* __restrict__ Whh2,
                          const float* __restrict__ bih2, const float* __restrict__ bhh2,
                          unsigned* __restrict__ apack) {
  int idx = blockIdx.x * blockDim.x + threadIdx.x;          // 0 .. 26*32*8-1
  if (idx >= NFRAG * 32 * 8) return;
  int slot = idx & 7;
  int lane = (idx >> 3) & 31;
  int f    = idx >> 8;
  int j = f >> 1, k = f & 1;
  int gp  = 16 * j + (lane & 15);                           // gate' row (0..207)
  int hi8 = (lane >= 16) ? 8 : 0;
  float vals[2];
#pragma unroll
  for (int p = 0; p < 2; ++p) {
    int Kr = 2 * slot + 2 * hi8 + p;
    int K  = 32 * k + Kr;                                   // 0..63
    float w = 0.0f;
    if (gp < 4 * Hn) {                                      // layer-1 gate rows
      int g = gp & 3, h = gp >> 2;                          // gate' = 4h+g
      int r = g * Hn + h;                                   // torch row (i,f,g,o blocks)
      if      (K < Hn)   w = Whh1[r * Hn + K];
      else if (K == 51)  w = Wih1[r];                       // x_t column
      else if (K == 53)  w = bih1[r] + bhh1[r];             // bias row (B[53]=1)
    } else {                                                // layer-2 gate rows
      int g2 = gp - 4 * Hn;                                 // 0..3
      if      (K < Hn)   w = Wih2[g2 * Hn + K];
      else if (K == 52)  w = Whh2[g2];                      // h2 recurrence
      else if (K == 53)  w = bih2[g2] + bhh2[g2];
    }
    vals[p] = w;
  }
  apack[idx] = pkh2(vals[0], vals[1]);
}

// ---------------------------------------------------------------------------
// Main recurrent kernel: 64 blocks x 32 threads (1 wave = one 16-row tile).
// D-fragment: lane holds (batch = lane%16), VGPR r -> gate' = 16j + r + 8*hi.
// => gate type = r&3 (uniform), h = 4j + 2*hi + (r>>2). Lane owns 26 state
// slots; pair q (h=2q,2q+1): q even owned by lanes<16, q odd by lanes>=16.
// B (32x16 f16) assumed: lanes 0-15 hold K=0..15 (2/VGPR), lanes 16-31 K=16..31.
// ---------------------------------------------------------------------------
__global__ void __launch_bounds__(32, 1)
lstm_main(const float* __restrict__ stim, const unsigned* __restrict__ apack,
          float* __restrict__ out) {
  const int lane = threadIdx.x;
  const bool hi  = lane >= 16;
  const int row  = blockIdx.x * 16 + (lane & 15);
  const float* xptr = stim + (size_t)row * Tn;

  // Resident A fragments (26 x v16h = 208 VGPRs).
  v16h A[NFRAG];
#pragma unroll
  for (int f = 0; f < NFRAG; ++f) {
    v8u t = *(const v8u*)(apack + (f * 32 + lane) * 8);
    A[f] = __builtin_bit_cast(v16h, t);
  }

  float cst[26] = {};      // cell state per owned (h,batch) slot; slot 25 on
  float hst[26] = {};      // hi lanes of tile 12 is layer-2 (c2/h2).
  unsigned hpk[NTILE] = {};// f16x2-packed owned h pairs (B building blocks)

  float xv = xptr[0];

  for (int t = 0; t <= Tn; ++t) {           // T+1 iters: layer2 retimed by 1
    int tn = (t + 1 < Tn) ? (t + 1) : (Tn - 1);
    float xnx = xptr[tn];                   // prefetch next x (latency hidden)

    // ---- build B fragments from previous-step state -----------------------
    unsigned sw[NTILE];
#pragma unroll
    for (int q = 0; q < NTILE; ++q) sw[q] = __shfl_xor(hpk[q], 16, 32);
    unsigned ux  = pkh2(hst[24], xv);       // (h50, x_t)   -> K 50,51 (hi lanes)
    unsigned uh2 = pkh2(hst[25], 1.0f);     // (h2, 1.0)    -> K 52,53 (hi lanes)

    v8u b0v, b1v;
#pragma unroll
    for (int v = 0; v < 4; ++v) {
      b0v[2 * v]     = hi ? sw[4 + v]  : hpk[v];
      b0v[2 * v + 1] = hi ? hpk[4 + v] : sw[v];
    }
    b1v[0] = hi ? sw[12] : hpk[8];
    b1v[1] = hi ? ux     : sw[8];
    b1v[2] = hi ? uh2    : hpk[9];
    b1v[3] = hi ? 0u     : sw[9];
    b1v[4] = hi ? 0u     : hpk[10];
    b1v[5] = hi ? 0u     : sw[10];
    b1v[6] = hi ? 0u     : hpk[11];
    b1v[7] = hi ? 0u     : sw[11];
    v16h B0 = __builtin_bit_cast(v16h, b0v);
    v16h B1 = __builtin_bit_cast(v16h, b1v);

    // ---- gates = A*B (bias folded in, C starts at 0), then pointwise ------
#pragma unroll
    for (int j = 0; j < NTILE; ++j) {
      v8f acc = {};
      acc = __builtin_amdgcn_wmma_f32_16x16x32_f16(false, A[2 * j],     false, B0,
                                                   (short)0, acc, false, false);
      acc = __builtin_amdgcn_wmma_f32_16x16x32_f16(false, A[2 * j + 1], false, B1,
                                                   (short)0, acc, false, false);
#pragma unroll
      for (int half = 0; half < 2; ++half) {
        const int r0 = 4 * half, idx = 2 * j + half;
        float si = sig_fast (acc[r0 + 0]);
        float sf = sig_fast (acc[r0 + 1]);
        float tg = tanh_fast(acc[r0 + 2]);
        float so = sig_fast (acc[r0 + 3]);
        float c  = __builtin_fmaf(sf, cst[idx], si * tg);
        cst[idx] = c;
        hst[idx] = so * tanh_fast(c);
        if (idx == 25) {                    // layer-2 slot (hi lanes only):
          if (t == 0 && hi) { cst[25] = 0.0f; hst[25] = 0.0f; }  // h2(-1)=0
        }
      }
    }

    // y[t-1] = h2(t-1), produced this iteration by the retimed layer 2.
    if (hi && t > 0) out[(size_t)row * Tn + (t - 1)] = hst[25];

    // repack h for next step's B
#pragma unroll
    for (int q = 0; q < NTILE; ++q) hpk[q] = pkh2(hst[2 * q], hst[2 * q + 1]);
    xv = xnx;
  }
}

// ---------------------------------------------------------------------------
extern "C" void kernel_launch(void* const* d_in, const int* in_sizes, int n_in,
                              void* d_out, int out_size, void* d_ws, size_t ws_size,
                              hipStream_t stream) {
  const float* stim = (const float*)d_in[0];
  const float* Wih1 = (const float*)d_in[1];
  const float* Whh1 = (const float*)d_in[2];
  const float* bih1 = (const float*)d_in[3];
  const float* bhh1 = (const float*)d_in[4];
  const float* Wih2 = (const float*)d_in[5];
  const float* Whh2 = (const float*)d_in[6];
  const float* bih2 = (const float*)d_in[7];
  const float* bhh2 = (const float*)d_in[8];
  unsigned* apack = (unsigned*)d_ws;        // 26*32*8*4 = 26.6 KB scratch

  lstm_pack<<<NFRAG, 256, 0, stream>>>(Wih1, Whh1, bih1, bhh1,
                                       Wih2, Whh2, bih2, bhh2, apack);
  lstm_main<<<Nn / 16, 32, 0, stream>>>(stim, apack, (float*)d_out);
}
